// IdentityAwaredCalibModule_30253749633332
// MI455X (gfx1250) — compile-verified, matrix operand order checked by reading, not run on hardware
//
#include <hip/hip_runtime.h>

// Problem constants (from reference): T=512, N=512, D=256, H=128, 4H=512, OUT=8
#define T_STEPS 512
#define N_BATCH 512
#define D_IN    256
#define H_DIM   128
#define G_DIM   512
#define OUT_DIM 8

typedef __attribute__((ext_vector_type(16))) __bf16 v16bf;
typedef __attribute__((ext_vector_type(8)))  __bf16 v8bf;
typedef __attribute__((ext_vector_type(8)))  float  v8f;
typedef __attribute__((ext_vector_type(4)))  float  v4f;

static __device__ inline v8f zero8() {
  v8f z;
#pragma unroll
  for (int i = 0; i < 8; ++i) z[i] = 0.f;
  return z;
}
static __device__ inline float sigmoidf_(float x) { return 1.f / (1.f + __expf(-x)); }

// ---------------------------------------------------------------------------
// LDS layout (dynamic shared):
//   [0      , 262144)  W bf16 B-fragments: 32 col-tiles x 8 K-chunks x 32 lanes x 16 bf16
//   [262144 , 266240)  h buffer: 16 rows x 128 cols bf16, row-major
//   [266240 , 270336)  calib_w fp32 [8][128]
// Total 270336 bytes (<= 320 KB WGP LDS).
// ---------------------------------------------------------------------------
#define WFRAG_OFF 0
#define HBUF_OFF  262144
#define CW_OFF    266240
#define SMEM_BYTES 270336

// Fill LDS with bf16 B-fragments of W.T where W is [G_DIM][256] fp32 row-major.
// B = W.T is [K=256][G_DIM]. Column tile ct = q*8 + w covers gate cols
// g = q*128 + w*16 + n (n = lane&15). Fragment mapping (16x16x32 bf16 B):
//   lanes 0-15 : n = lane,    K = kb..kb+7 (slots 0-7), kb+16..kb+23 (slots 8-15)
//   lanes 16-31: n = lane-16, kb += 8
static __device__ void fill_w_frags(const float* __restrict__ W,
                                    __bf16* __restrict__ wfrag, int tid) {
  for (int idx = tid; idx < 32 * 8 * 32; idx += 256) {
    int lane = idx & 31;
    int kc   = (idx >> 5) & 7;
    int ct   = idx >> 8;
    int q = ct >> 3, w = ct & 7;
    int g  = q * 128 + w * 16 + (lane & 15);
    int kb = kc * 32 + ((lane & 16) ? 8 : 0);
    __bf16* dst = wfrag + ((ct * 8 + kc) * 32 + lane) * 16;
    const float* src = W + (size_t)g * D_IN;
#pragma unroll
    for (int s = 0; s < 16; ++s) {
      int k = kb + ((s < 8) ? s : (8 + s));
      dst[s] = (__bf16)src[k];
    }
  }
}

// A-fragment (16x32 bf16) from fp32 row-major global memory (rowStride floats).
static __device__ inline v16bf load_a_frag_f32(const float* __restrict__ base,
                                               int rowStride, int kc, int lane) {
  int m  = lane & 15;
  int kb = kc * 32 + ((lane & 16) ? 8 : 0);
  const float* p = base + (size_t)m * rowStride + kb;
  v4f a0 = *(const v4f*)(p);
  v4f a1 = *(const v4f*)(p + 4);
  v4f a2 = *(const v4f*)(p + 16);
  v4f a3 = *(const v4f*)(p + 20);
  v16bf u;
#pragma unroll
  for (int s = 0; s < 4; ++s) {
    u[s]      = (__bf16)a0[s];
    u[4 + s]  = (__bf16)a1[s];
    u[8 + s]  = (__bf16)a2[s];
    u[12 + s] = (__bf16)a3[s];
  }
  return u;
}

// A-fragment from bf16 row-major [16][128] LDS buffer (two ds_load_b128 per lane).
static __device__ inline v16bf load_a_frag_lds(const __bf16* __restrict__ hbuf,
                                               int kc, int lane) {
  int m  = lane & 15;
  int kb = kc * 32 + ((lane & 16) ? 8 : 0);
  const __bf16* p = hbuf + m * H_DIM + kb;
  v8bf lo = *(const v8bf*)(p);
  v8bf hi = *(const v8bf*)(p + 16);
  return __builtin_shufflevector(lo, hi, 0, 1, 2, 3, 4, 5, 6, 7,
                                 8, 9, 10, 11, 12, 13, 14, 15);
}

// B-fragment from the LDS fragment area (one 32-byte contiguous read per lane).
static __device__ inline v16bf load_b_frag(const __bf16* __restrict__ wfrag,
                                           int ct, int kc, int lane) {
  const __bf16* p = wfrag + ((ct * 8 + kc) * 32 + lane) * 16;
  v8bf lo = *(const v8bf*)(p);
  v8bf hi = *(const v8bf*)(p + 8);
  return __builtin_shufflevector(lo, hi, 0, 1, 2, 3, 4, 5, 6, 7,
                                 8, 9, 10, 11, 12, 13, 14, 15);
}

#define WMMA_BF16(A, B, C)                                                     \
  __builtin_amdgcn_wmma_f32_16x16x32_bf16(false, (A), false, (B), (short)0,    \
                                          (C), false, false)

__global__ void __launch_bounds__(256, 1)
lstm_calib_kernel(const float* __restrict__ x,        // [T,N,D]
                  const float* __restrict__ identity, // [N,D]
                  const float* __restrict__ wih,      // [4H,D]
                  const float* __restrict__ whh,      // [4H,H]
                  const float* __restrict__ wii,      // [4H,D]
                  const float* __restrict__ bih,      // [4H]
                  const float* __restrict__ bhh,      // [4H]
                  const float* __restrict__ bii,      // [4H]
                  const float* __restrict__ cw,       // [8,H]
                  const float* __restrict__ cb,       // [8]
                  float* __restrict__ out)            // [N,T,8]
{
  extern __shared__ unsigned char smem[];
  __bf16* wfrag = (__bf16*)(smem + WFRAG_OFF);
  __bf16* hbuf  = (__bf16*)(smem + HBUF_OFF);
  float*  cwl   = (float*)(smem + CW_OFF);

  const int tid  = threadIdx.x;
  const int lane = tid & 31;
  const int w    = tid >> 5;          // wave id 0..7 -> hidden cols 16w..16w+15
  const int n0   = blockIdx.x * 16;   // this WG owns batch rows n0..n0+15

  // One-time LDS init: calib weights + zero h(t=-1)
  for (int i = tid; i < OUT_DIM * H_DIM; i += 256) cwl[i] = cw[i];
  for (int i = tid; i < 16 * H_DIM; i += 256) hbuf[i] = (__bf16)0.f;

  // ---- Phase 1: igates = identity @ W_ii.T + (b_ii + b_ih + b_hh), in-regs ----
  fill_w_frags(wii, wfrag, tid);
  __syncthreads();

  v8f iacc[4];
#pragma unroll
  for (int q = 0; q < 4; ++q) iacc[q] = zero8();
#pragma unroll
  for (int kc = 0; kc < 8; ++kc) {
    v16bf a = load_a_frag_f32(identity + (size_t)n0 * D_IN, D_IN, kc, lane);
#pragma unroll
    for (int q = 0; q < 4; ++q) {
      v16bf b = load_b_frag(wfrag, q * 8 + w, kc, lane);
      iacc[q] = WMMA_BF16(a, b, iacc[q]);
    }
  }
  {
    const int col0 = w * 16 + (lane & 15);
#pragma unroll
    for (int q = 0; q < 4; ++q) {
      int g = q * 128 + col0;
      float bsum = bii[g] + bih[g] + bhh[g];
#pragma unroll
      for (int v = 0; v < 8; ++v) iacc[q][v] += bsum;
    }
  }
  __syncthreads();

  // ---- Phase 2: stage W_ih into LDS; W_hh B-fragments into VGPRs ----
  fill_w_frags(wih, wfrag, tid);

  v16bf Bhh[4][4];
#pragma unroll
  for (int q = 0; q < 4; ++q) {
    int g = q * 128 + w * 16 + (lane & 15);
    const float* src = whh + (size_t)g * H_DIM;
#pragma unroll
    for (int kc = 0; kc < 4; ++kc) {
      int kb = kc * 32 + ((lane & 16) ? 8 : 0);
      v16bf u;
#pragma unroll
      for (int s = 0; s < 16; ++s) {
        int k = kb + ((s < 8) ? s : (8 + s));
        u[s] = (__bf16)src[k];
      }
      Bhh[q][kc] = u;
    }
  }

  v8f creg = zero8();                 // cell state tile, fp32 in regs
  const int prow = tid >> 3;          // projection: row (tid<128)
  const int pout = tid & 7;           // projection: output channel
  const float pcb = (tid < 128) ? cb[pout] : 0.f;

  __syncthreads();

  // ---- Recurrent T-loop: fully WG-local, no grid sync needed ----
  for (int t = 0; t < T_STEPS; ++t) {
    // h(t-1) A-fragments from LDS
    v16bf ah[4];
#pragma unroll
    for (int kc = 0; kc < 4; ++kc) ah[kc] = load_a_frag_lds(hbuf, kc, lane);

    v8f acc[4];
#pragma unroll
    for (int q = 0; q < 4; ++q) acc[q] = iacc[q];

    const float* xbase = x + ((size_t)t * N_BATCH + n0) * D_IN;
    if (t + 1 < T_STEPS)  // pull next step's x rows toward the WGP
      __builtin_prefetch(x + ((size_t)(t + 1) * N_BATCH + n0 + (lane & 15)) * D_IN, 0, 1);

    // gates += x_t @ W_ih.T   (K = 256)
#pragma unroll
    for (int kc = 0; kc < 8; ++kc) {
      v16bf a = load_a_frag_f32(xbase, D_IN, kc, lane);
#pragma unroll
      for (int q = 0; q < 4; ++q) {
        v16bf b = load_b_frag(wfrag, q * 8 + w, kc, lane);
        acc[q] = WMMA_BF16(a, b, acc[q]);
      }
    }
    // gates += h_{t-1} @ W_hh.T   (K = 128, weights in VGPRs)
#pragma unroll
    for (int kc = 0; kc < 4; ++kc) {
#pragma unroll
      for (int q = 0; q < 4; ++q) acc[q] = WMMA_BF16(ah[kc], Bhh[q][kc], acc[q]);
    }

    // LSTM pointwise update (i,f,g,o share lane layout by construction)
    float hv[8];
#pragma unroll
    for (int v = 0; v < 8; ++v) {
      float ig = sigmoidf_(acc[0][v]);
      float fg = sigmoidf_(acc[1][v]);
      float gg = tanhf(acc[2][v]);
      float og = sigmoidf_(acc[3][v]);
      float c  = fg * creg[v] + ig * gg;
      creg[v]  = c;
      hv[v]    = og * tanhf(c);
    }

    __syncthreads();   // everyone done reading old h
    {
      // C/D layout: lanes 0-15 -> col=lane, row=v ; lanes 16-31 -> row=8+v
      int col   = w * 16 + (lane & 15);
      int mbase = (lane < 16) ? 0 : 8;
#pragma unroll
      for (int v = 0; v < 8; ++v)
        hbuf[(mbase + v) * H_DIM + col] = (__bf16)hv[v];
    }
    __syncthreads();   // new h visible to all waves

    // y[n, t, :] = h_t[n, :] @ calib_w.T + calib_b   (tiny: 16x8, K=128)
    if (tid < 128) {
      const float* cwrow = cwl + pout * H_DIM;
      const __bf16* hrow = hbuf + prow * H_DIM;
      float acc2 = pcb;
#pragma unroll 8
      for (int k = 0; k < H_DIM; ++k) acc2 += (float)hrow[k] * cwrow[k];
      out[((size_t)(n0 + prow) * T_STEPS + t) * OUT_DIM + pout] = acc2;
    }
    // next iteration's reads of hbuf are protected by the barrier above;
    // its next overwrite is protected by the pre-write barrier.
  }
}

extern "C" void kernel_launch(void* const* d_in, const int* in_sizes, int n_in,
                              void* d_out, int out_size, void* d_ws, size_t ws_size,
                              hipStream_t stream) {
  const float* x    = (const float*)d_in[0];
  const float* iden = (const float*)d_in[1];
  const float* wih  = (const float*)d_in[2];
  const float* whh  = (const float*)d_in[3];
  const float* wii  = (const float*)d_in[4];
  const float* bih  = (const float*)d_in[5];
  const float* bhh  = (const float*)d_in[6];
  const float* bii  = (const float*)d_in[7];
  const float* cw   = (const float*)d_in[8];
  const float* cb   = (const float*)d_in[9];
  float* out = (float*)d_out;

  (void)d_ws; (void)ws_size; (void)in_sizes; (void)n_in; (void)out_size;

  // 270336 B of dynamic LDS (> 64 KB default): opt in every call (deterministic).
  hipFuncSetAttribute((const void*)lstm_calib_kernel,
                      hipFuncAttributeMaxDynamicSharedMemorySize, SMEM_BYTES);

  lstm_calib_kernel<<<dim3(N_BATCH / 16), dim3(256), SMEM_BYTES, stream>>>(
      x, iden, wih, whh, wii, bih, bhh, bii, cw, cb, out);
}
